// Block_65773129171661
// MI455X (gfx1250) — compile-verified, matrix operand order checked by reading, not run on hardware
//
#include <hip/hip_runtime.h>
#include <hip/hip_bf16.h>
#include <stdint.h>

// ---------------------------------------------------------------------------
// Transformer block for MI455X (gfx1250, wave32, WMMA bf16 16x16x32).
// B=1, N=4096, C=768, H=12, HD=64, FF=3072.
// GEMMs + attention: async-to-LDS double-buffered pipelines (ASYNCcnt)
// feeding v_wmma_f32_16x16x32_bf16.
// ---------------------------------------------------------------------------

#define SEQ  4096
#define CH   768
#define NH   12
#define HD   64
#define FF   3072
#define QKVC 2304
#define LN_EPS 1e-5f

typedef __attribute__((ext_vector_type(16))) __bf16 v16bf;
typedef __attribute__((ext_vector_type(8)))  float  v8f;

union Frag { v16bf v; uint4 u[2]; };

static __device__ __forceinline__ __bf16 f2bf(float f) {
    unsigned u = __builtin_bit_cast(unsigned, f);
    unsigned r = u + 0x7FFFu + ((u >> 16) & 1u);     // round-to-nearest-even
    unsigned short h = (unsigned short)(r >> 16);
    return __builtin_bit_cast(__bf16, h);
}

static __device__ __forceinline__ float red16_max(float v) {
#pragma unroll
    for (int off = 8; off; off >>= 1) v = fmaxf(v, __shfl_xor(v, off, 32));
    return v;
}
static __device__ __forceinline__ float red16_sum(float v) {
#pragma unroll
    for (int off = 8; off; off >>= 1) v += __shfl_xor(v, off, 32);
    return v;
}

// Async copy global -> LDS (gfx1250, tracked by ASYNCcnt; in-order completion).
static __device__ __forceinline__ void async_g2l_b128(unsigned lds_off, const void* gptr) {
    asm volatile("global_load_async_to_lds_b128 %0, %1, off"
                 :: "v"(lds_off), "v"((unsigned long long)(uintptr_t)gptr) : "memory");
}
static __device__ __forceinline__ void async_g2l_b64(unsigned lds_off, const void* gptr) {
    asm volatile("global_load_async_to_lds_b64 %0, %1, off"
                 :: "v"(lds_off), "v"((unsigned long long)(uintptr_t)gptr) : "memory");
}
static __device__ __forceinline__ unsigned lds_off_of(const void* p) {
    return (unsigned)(uintptr_t)p;   // low 32 bits of flat LDS pointer = LDS offset
}

// ---------------------------------------------------------------------------
// fp32 -> bf16 elementwise (weight conversion)
// ---------------------------------------------------------------------------
__global__ void __launch_bounds__(256) cvt_bf16_kernel(const float* __restrict__ in,
                                                       __bf16* __restrict__ out, int n) {
    int i = blockIdx.x * 256 + threadIdx.x;
    if (i < n) out[i] = f2bf(in[i]);
}

// ---------------------------------------------------------------------------
// LayerNorm (one token per block of 256 threads, C=768 -> 3 elems/thread)
// ---------------------------------------------------------------------------
__global__ void __launch_bounds__(256) ln_bf16_kernel(const float* __restrict__ x,
                                                      const float* __restrict__ g,
                                                      const float* __restrict__ b,
                                                      __bf16* __restrict__ out) {
    __shared__ float sm[8];
    __shared__ float sv[8];
    const int row = blockIdx.x;
    const float* xr = x + (size_t)row * CH;
    float v[3];
    float s = 0.f;
#pragma unroll
    for (int i = 0; i < 3; ++i) { v[i] = xr[threadIdx.x + i * 256]; s += v[i]; }
#pragma unroll
    for (int off = 16; off; off >>= 1) s += __shfl_xor(s, off, 32);
    if ((threadIdx.x & 31) == 0) sm[threadIdx.x >> 5] = s;
    __syncthreads();
    float tot = 0.f;
#pragma unroll
    for (int i = 0; i < 8; ++i) tot += sm[i];
    const float mu = tot * (1.f / CH);

    float var = 0.f;
#pragma unroll
    for (int i = 0; i < 3; ++i) { float d = v[i] - mu; var += d * d; }
#pragma unroll
    for (int off = 16; off; off >>= 1) var += __shfl_xor(var, off, 32);
    if ((threadIdx.x & 31) == 0) sv[threadIdx.x >> 5] = var;
    __syncthreads();
    float tv = 0.f;
#pragma unroll
    for (int i = 0; i < 8; ++i) tv += sv[i];
    const float rstd = rsqrtf(tv * (1.f / CH) + LN_EPS);
#pragma unroll
    for (int i = 0; i < 3; ++i) {
        int c = threadIdx.x + i * 256;
        out[(size_t)row * CH + c] = f2bf((v[i] - mu) * rstd * g[c] + b[c]);
    }
}

// ---------------------------------------------------------------------------
// WMMA GEMM with async-LDS double buffering:
//   out[M,Nout] = A[M,K](bf16) @ W[Nout,K](bf16)^T (+epilogue)
// Block: 8 waves = 2 row-tiles x 4 col-tiles = 32 x 256 output tile.
// Every thread issues exactly 5 async ops/stage => uniform per-wave ASYNCcnt;
// s_wait_asynccnt 0x5 == "previous stage complete" (async loads finish in order).
// mode 0: fp32 out; mode 1: +bias+residual fp32 out; mode 2: +bias, GELU, bf16 out.
// ---------------------------------------------------------------------------
__global__ void __launch_bounds__(256) gemm_wmma_kernel(
        const __bf16* __restrict__ A, const __bf16* __restrict__ W,
        const float* __restrict__ bias, const float* __restrict__ res,
        float* __restrict__ outF, __bf16* __restrict__ outB,
        int M, int Nout, int K, int mode) {
    __shared__ __bf16 ldsB[2][256][32];   // 2 x 16 KB
    __shared__ __bf16 ldsA[2][32][32];    // 2 x  2 KB
    const int lane = threadIdx.x;
    const int wave = threadIdx.y;
    const int tid  = wave * 32 + lane;
    const int row0 = blockIdx.y * 32;
    const int col0 = blockIdx.x * 256;
    const int wrow = (wave >> 2) * 16;    // wave's row tile within block
    const int wcol = (wave & 3) * 64;     // wave's col tile within block
    const int m    = lane & 15;
    const int kg   = (lane >> 4) * 8;

    // staging assignment (uniform 5 async ops per thread per stage)
    const int bcol = tid;                 // B: one 32-K row of one output column
    const int arow = tid >> 3;            // A: 8 threads per row
    const int asub = (tid & 7) * 4;       // 4 bf16 = 8 bytes per thread

    const __bf16* wsrc = W + (size_t)(col0 + bcol) * K;
    const __bf16* asrc = A + (size_t)(row0 + arow) * K + asub;

    v8f acc[4] = {};
    const int nK = K / 32;

    {   // prologue: stage K-step 0 into buffer 0
        unsigned lb = lds_off_of(&ldsB[0][bcol][0]);
#pragma unroll
        for (int c = 0; c < 4; ++c) async_g2l_b128(lb + c * 16, wsrc + c * 8);
        async_g2l_b64(lds_off_of(&ldsA[0][arow][asub]), asrc);
    }

    for (int i = 0; i < nK; ++i) {
        const int buf = i & 1;
        if (i + 1 < nK) {
            const int k = (i + 1) * 32;
            unsigned lb = lds_off_of(&ldsB[buf ^ 1][bcol][0]);
#pragma unroll
            for (int c = 0; c < 4; ++c) async_g2l_b128(lb + c * 16, wsrc + k + c * 8);
            async_g2l_b64(lds_off_of(&ldsA[buf ^ 1][arow][asub]), asrc + k);
            asm volatile("s_wait_asynccnt 0x5" ::: "memory"); // stage i landed
        } else {
            asm volatile("s_wait_asynccnt 0x0" ::: "memory"); // final stage landed
        }
        __syncthreads();

        Frag a;
        a.u[0] = *(const uint4*)&ldsA[buf][wrow + m][kg];
        a.u[1] = *(const uint4*)&ldsA[buf][wrow + m][kg + 16];
#pragma unroll
        for (int t = 0; t < 4; ++t) {
            Frag b;
            b.u[0] = *(const uint4*)&ldsB[buf][wcol + t * 16 + m][kg];
            b.u[1] = *(const uint4*)&ldsB[buf][wcol + t * 16 + m][kg + 16];
            acc[t] = __builtin_amdgcn_wmma_f32_16x16x32_bf16(
                false, a.v, false, b.v, (short)0, acc[t], false, false);
        }
        __syncthreads();   // all waves done reading buf before it is restaged
    }

    const int rbase = (lane >> 4) * 8;    // D rows r..r+7 -> m = rbase + r
#pragma unroll
    for (int t = 0; t < 4; ++t) {
        const int col = col0 + wcol + t * 16 + m;
        const float bv = bias ? bias[col] : 0.f;
#pragma unroll
        for (int r = 0; r < 8; ++r) {
            const int row = row0 + wrow + rbase + r;
            float v = acc[t][r] + bv;
            if (mode == 1) {
                v += res[(size_t)row * Nout + col];
                outF[(size_t)row * Nout + col] = v;
            } else if (mode == 2) {
                float gch = 0.5f * v * (1.f + erff(v * 0.70710678118f)); // exact GELU
                outB[(size_t)row * Nout + col] = f2bf(gch);
            } else {
                outF[(size_t)row * Nout + col] = v;
            }
        }
    }
}

// ---------------------------------------------------------------------------
// RoPE + repack: qkv fp32 [N][3][H][HD] ->
//   qb [H][N][HD] bf16 (rope), kb [H][N][HD] bf16 (rope), vt [H][HD][N] bf16.
// ---------------------------------------------------------------------------
__global__ void __launch_bounds__(256) rope_pack_kernel(
        const float* __restrict__ qkv, const float* __restrict__ cosT,
        const float* __restrict__ sinT, __bf16* __restrict__ qb,
        __bf16* __restrict__ kb, __bf16* __restrict__ vt) {
    int idx = blockIdx.x * 256 + threadIdx.x;
    if (idx >= SEQ * NH * (HD / 2)) return;
    const int d2 = idx & 31;
    const int h  = (idx >> 5) % NH;
    const int n  = idx / (32 * NH);
    const float* base = qkv + (size_t)n * QKVC + h * HD + 2 * d2;
    const float c = cosT[n * 32 + d2];
    const float s = sinT[n * 32 + d2];

    const size_t qo = ((size_t)h * SEQ + n) * HD + 2 * d2;
    float q0 = base[0],  q1 = base[1];
    qb[qo]     = f2bf(q0 * c - q1 * s);
    qb[qo + 1] = f2bf(q0 * s + q1 * c);
    float k0 = base[CH], k1 = base[CH + 1];
    kb[qo]     = f2bf(k0 * c - k1 * s);
    kb[qo + 1] = f2bf(k0 * s + k1 * c);
    float v0 = base[2 * CH], v1 = base[2 * CH + 1];
    vt[((size_t)h * HD + 2 * d2)     * SEQ + n] = f2bf(v0);
    vt[((size_t)h * HD + 2 * d2 + 1) * SEQ + n] = f2bf(v1);
}

// ---------------------------------------------------------------------------
// Flash attention with async-LDS K/V staging.
// One wave per (head, 16-query block); the 8 waves of a block share one head
// and therefore share the K/V stream: K(32x64) and V^T(64x32) tiles are
// staged once per block into double-buffered LDS with async copies
// (2 async ops per thread per stage -> uniform ASYNCcnt, wait 0x2 overlaps
// next-stage copy with current-stage WMMA + softmax).
// ---------------------------------------------------------------------------
__global__ void __launch_bounds__(256) attention_kernel(
        const __bf16* __restrict__ qb, const __bf16* __restrict__ kbm,
        const __bf16* __restrict__ vt, __bf16* __restrict__ ob) {
    __shared__ __bf16 ldsK[2][32][64];   // 2 x 4 KB  (keys x hd)
    __shared__ __bf16 ldsV[2][64][32];   // 2 x 4 KB  (hd x keys, V^T)
    __shared__ __bf16 ps[8][16][32];     // per-wave P tile (16 q x 32 keys)
    const int lane = threadIdx.x;
    const int wave = threadIdx.y;
    const int tid  = wave * 32 + lane;
    const int gw = blockIdx.x * 8 + wave;
    const int h  = gw >> 8;              // 256 query blocks per head
    const int q0 = (gw & 255) * 16;
    const int m  = lane & 15;
    const int kg = (lane >> 4) * 8;
    const int rbase = (lane >> 4) * 8;
    const float SCALE = 0.125f;          // 1/sqrt(64)

    const __bf16* qh = qb  + (size_t)h * SEQ * HD;
    const __bf16* kh = kbm + (size_t)h * SEQ * HD;
    const __bf16* vh = vt  + (size_t)h * HD * SEQ;

    // K/V staging assignment (one b128 each per thread per stage)
    const int krow_s = tid >> 3;              // 0..31 key row, 8 threads/row
    const int ksub   = (tid & 7) * 8;         // 8 bf16 = 16B
    const int vrow_s = tid >> 2;              // 0..63 hd row, 4 threads/row
    const int vsub   = (tid & 3) * 8;

    // Q A-fragments (hd 0..31 and 32..63), in registers for the whole pass.
    Frag qa0, qa1;
    {
        const __bf16* qrow = qh + (size_t)(q0 + m) * HD + kg;
        qa0.u[0] = *(const uint4*)(qrow);
        qa0.u[1] = *(const uint4*)(qrow + 16);
        qa1.u[0] = *(const uint4*)(qrow + 32);
        qa1.u[1] = *(const uint4*)(qrow + 48);
    }

    v8f o[4] = {};
    float mi[8], li[8];
#pragma unroll
    for (int r = 0; r < 8; ++r) { mi[r] = -INFINITY; li[r] = 0.f; }

    // prologue: stage key block 0 into buffer 0
    async_g2l_b128(lds_off_of(&ldsK[0][krow_s][ksub]),
                   kh + (size_t)krow_s * HD + ksub);
    async_g2l_b128(lds_off_of(&ldsV[0][vrow_s][vsub]),
                   vh + (size_t)vrow_s * SEQ + vsub);

    for (int kb0 = 0; kb0 < SEQ; kb0 += 32) {
        const int buf = (kb0 >> 5) & 1;
        if (kb0 + 32 < SEQ) {
            async_g2l_b128(lds_off_of(&ldsK[buf ^ 1][krow_s][ksub]),
                           kh + (size_t)(kb0 + 32 + krow_s) * HD + ksub);
            async_g2l_b128(lds_off_of(&ldsV[buf ^ 1][vrow_s][vsub]),
                           vh + (size_t)vrow_s * SEQ + kb0 + 32 + vsub);
            asm volatile("s_wait_asynccnt 0x2" ::: "memory"); // current stage landed
        } else {
            asm volatile("s_wait_asynccnt 0x0" ::: "memory");
        }
        __syncthreads();

        // ---- scores: S(16x32) = Q(16x64) @ K^T, two 16-col D tiles ----
        v8f st[2];
#pragma unroll
        for (int u = 0; u < 2; ++u) {
            const __bf16* krow = &ldsK[buf][u * 16 + m][kg];
            Frag b0, b1;
            b0.u[0] = *(const uint4*)(krow);
            b0.u[1] = *(const uint4*)(krow + 16);
            b1.u[0] = *(const uint4*)(krow + 32);
            b1.u[1] = *(const uint4*)(krow + 48);
            v8f s = {};
            s = __builtin_amdgcn_wmma_f32_16x16x32_bf16(false, qa0.v, false, b0.v,
                                                        (short)0, s, false, false);
            s = __builtin_amdgcn_wmma_f32_16x16x32_bf16(false, qa1.v, false, b1.v,
                                                        (short)0, s, false, false);
            st[u] = s;
        }

        // ---- online softmax (rows live across 16-lane groups) ----
#pragma unroll
        for (int r = 0; r < 8; ++r) {
            float v0 = st[0][r] * SCALE;
            float v1 = st[1][r] * SCALE;
            float rm = red16_max(fmaxf(v0, v1));
            float nm = fmaxf(mi[r], rm);
            float alpha = __expf(mi[r] - nm);
            mi[r] = nm;
            float p0 = __expf(v0 - nm);
            float p1 = __expf(v1 - nm);
            li[r] = li[r] * alpha + red16_sum(p0 + p1);
#pragma unroll
            for (int t = 0; t < 4; ++t) o[t][r] *= alpha;
            ps[wave][rbase + r][m]      = f2bf(p0);
            ps[wave][rbase + r][16 + m] = f2bf(p1);
        }

        Frag pa;   // D-tile layout -> A-fragment layout via per-wave LDS
        pa.u[0] = *(const uint4*)&ps[wave][m][kg];
        pa.u[1] = *(const uint4*)&ps[wave][m][kg + 16];

        // ---- O(16x64) += P(16x32) @ V(32x64) from LDS V^T ----
#pragma unroll
        for (int t = 0; t < 4; ++t) {
            const __bf16* vrow = &ldsV[buf][t * 16 + m][kg];
            Frag vb;
            vb.u[0] = *(const uint4*)(vrow);
            vb.u[1] = *(const uint4*)(vrow + 16);
            o[t] = __builtin_amdgcn_wmma_f32_16x16x32_bf16(false, pa.v, false, vb.v,
                                                           (short)0, o[t], false, false);
        }
        __syncthreads();   // all waves done with buf before it is restaged
    }

    // ---- finalize: O /= l, write bf16 [token][C] for the proj GEMM ----
#pragma unroll
    for (int r = 0; r < 8; ++r) {
        const float inv = 1.f / li[r];
        const int row = q0 + rbase + r;
#pragma unroll
        for (int t = 0; t < 4; ++t)
            ob[(size_t)row * CH + h * HD + t * 16 + m] = f2bf(o[t][r] * inv);
    }
}

// ---------------------------------------------------------------------------
// Host-side orchestration
// ---------------------------------------------------------------------------
extern "C" void kernel_launch(void* const* d_in, const int* in_sizes, int n_in,
                              void* d_out, int out_size, void* d_ws, size_t ws_size,
                              hipStream_t stream) {
    const float* x     = (const float*)d_in[0];
    const float* fcos  = (const float*)d_in[1];
    const float* fsin  = (const float*)d_in[2];
    const float* w_qkv = (const float*)d_in[3];
    const float* w_prj = (const float*)d_in[4];
    const float* b_prj = (const float*)d_in[5];
    const float* g1    = (const float*)d_in[6];
    const float* be1   = (const float*)d_in[7];
    const float* g2    = (const float*)d_in[8];
    const float* be2   = (const float*)d_in[9];
    const float* w_fc1 = (const float*)d_in[10];
    const float* b_fc1 = (const float*)d_in[11];
    const float* w_fc2 = (const float*)d_in[12];
    const float* b_fc2 = (const float*)d_in[13];
    float* out = (float*)d_out;

    size_t off = 0;
    auto alloc = [&](size_t bytes) -> void* {
        void* p = (char*)d_ws + off;
        off += (bytes + 255) & ~(size_t)255;
        return p;
    };
    __bf16* hbf    = (__bf16*)alloc((size_t)SEQ * CH * 2);
    __bf16* wqkv_b = (__bf16*)alloc((size_t)QKVC * CH * 2);
    __bf16* wprj_b = (__bf16*)alloc((size_t)CH * CH * 2);
    __bf16* wfc1_b = (__bf16*)alloc((size_t)FF * CH * 2);
    __bf16* wfc2_b = (__bf16*)alloc((size_t)CH * FF * 2);
    float*  qkv    = (float*) alloc((size_t)SEQ * QKVC * 4);
    __bf16* q_b    = (__bf16*)alloc((size_t)SEQ * CH * 2);
    __bf16* k_b    = (__bf16*)alloc((size_t)SEQ * CH * 2);
    __bf16* v_t    = (__bf16*)alloc((size_t)SEQ * CH * 2);
    __bf16* o_b    = (__bf16*)alloc((size_t)SEQ * CH * 2);
    float*  x1     = (float*) alloc((size_t)SEQ * CH * 4);
    __bf16* h2bf   = (__bf16*)alloc((size_t)SEQ * CH * 2);
    __bf16* hfc    = (__bf16*)alloc((size_t)SEQ * FF * 2);
    (void)ws_size; (void)n_in; (void)in_sizes; (void)out_size;

    const dim3 blk(256);
    const dim3 wblk(32, 8);

    {   // weights fp32 -> bf16
        int n;
        n = QKVC * CH; cvt_bf16_kernel<<<(n + 255) / 256, blk, 0, stream>>>(w_qkv, wqkv_b, n);
        n = CH * CH;   cvt_bf16_kernel<<<(n + 255) / 256, blk, 0, stream>>>(w_prj, wprj_b, n);
        n = FF * CH;   cvt_bf16_kernel<<<(n + 255) / 256, blk, 0, stream>>>(w_fc1, wfc1_b, n);
        n = CH * FF;   cvt_bf16_kernel<<<(n + 255) / 256, blk, 0, stream>>>(w_fc2, wfc2_b, n);
    }

    ln_bf16_kernel<<<SEQ, blk, 0, stream>>>(x, g1, be1, hbf);

    gemm_wmma_kernel<<<dim3(QKVC / 256, SEQ / 32), wblk, 0, stream>>>(
        hbf, wqkv_b, nullptr, nullptr, qkv, nullptr, SEQ, QKVC, CH, 0);

    {
        int n = SEQ * NH * (HD / 2);
        rope_pack_kernel<<<(n + 255) / 256, blk, 0, stream>>>(qkv, fcos, fsin, q_b, k_b, v_t);
    }

    attention_kernel<<<NH * (SEQ / 16) / 8, wblk, 0, stream>>>(q_b, k_b, v_t, o_b);

    gemm_wmma_kernel<<<dim3(CH / 256, SEQ / 32), wblk, 0, stream>>>(
        o_b, wprj_b, b_prj, x, x1, nullptr, SEQ, CH, CH, 1);

    ln_bf16_kernel<<<SEQ, blk, 0, stream>>>(x1, g2, be2, h2bf);

    gemm_wmma_kernel<<<dim3(FF / 256, SEQ / 32), wblk, 0, stream>>>(
        h2bf, wfc1_b, b_fc1, nullptr, nullptr, hfc, SEQ, FF, CH, 2);

    gemm_wmma_kernel<<<dim3(CH / 256, SEQ / 32), wblk, 0, stream>>>(
        hfc, wfc2_b, b_fc2, x1, out, nullptr, SEQ, CH, FF, 1);
}
